// Model_12945031430945
// MI455X (gfx1250) — matrix operand order repaired and checked by reading, not compiled, not run on hardware
//
#include <hip/hip_runtime.h>

typedef _Float16 v16h __attribute__((ext_vector_type(16)));
typedef _Float16 v8h  __attribute__((ext_vector_type(8)));
typedef float    v8f  __attribute__((ext_vector_type(8)));

namespace {
constexpr int BN_ = 16;      // batch
constexpr int T_  = 4096;    // frames
constexpr int F_  = 80;      // mel bins
constexpr int C_  = 512;     // conv channels
constexpr int ZD_ = 64, ZN_ = 512, YD_ = 128;
constexpr int TP_ = T_ + 4;  // padded time rows
}

union HFrag { v16h v; v8h h[2]; };

// Async DMA global -> LDS (CDNA5, ASYNCcnt-tracked). lds32 = LDS byte address
// (low 32 bits of flat shared pointer, per ISA aperture rules).
#define ASYNC_B128(lds32, gp)                                                  \
    asm volatile("global_load_async_to_lds_b128 %0, %1, off" ::"v"(lds32),     \
                 "v"((unsigned long long)(uintptr_t)(gp))                      \
                 : "memory")
#define WAIT_ASYNC0() asm volatile("s_wait_asynccnt 0x0" ::: "memory")

// ---------------------------------------------------------------------------
// Conv-as-GEMM with WMMA f16: out[b,t,co] = sum_j A[b,t,j] * W[co,j] + bias[co]
// A row (b,t) = contiguous window  in + b*inBatch + t*sIn .. + Kdim
// Block tile: 128 (time) x 128 (cout); 8 waves as 4(M)x2(N), each wave 32x64
// (8 wmma accums). LDS double-buffered, filled by async DMA.
// ---------------------------------------------------------------------------
template <typename OutT>
__global__ __launch_bounds__(256)
void conv_gemm(const _Float16* __restrict__ in, size_t inBatch, int sIn,
               const _Float16* __restrict__ W, const float* __restrict__ bias,
               int Kdim, int Cout, OutT* __restrict__ out, int sOut)
{
    __shared__ _Float16 As[2][128 * 32];   // 8 KB per buffer
    __shared__ _Float16 Bs[2][128 * 32];   // 8 KB per buffer

    const int tid  = threadIdx.x;
    const int lane = tid & 31;
    const int wave = tid >> 5;
    const int wm   = wave & 3;   // 0..3 : 32-row slice
    const int wn   = wave >> 2;  // 0..1 : 64-col slice
    const int mrow = lane & 15;
    const int hsel = lane >> 4;

    const int t0 = blockIdx.x * 128;
    const int n0 = blockIdx.y * 128;
    const _Float16* inb = in + (size_t)blockIdx.z * inBatch;

    // cooperative staging: each thread moves 32B of A and 32B of B per tile
    const int srow = tid >> 1;          // 0..127
    const int sseg = (tid & 1) * 16;    // 0 or 16 halves

    // pre-zero B slots for out-of-range output channels (never overwritten)
    if (n0 + srow >= Cout) {
        v8h z;
#pragma unroll
        for (int q = 0; q < 8; ++q) z[q] = (_Float16)0.f;
#pragma unroll
        for (int bufi = 0; bufi < 2; ++bufi) {
            *(v8h*)&Bs[bufi][srow * 32 + sseg]     = z;
            *(v8h*)&Bs[bufi][srow * 32 + sseg + 8] = z;
        }
    }

    auto stage = [&](int kk, int bufi) {
        const _Float16* ga = inb + (size_t)(t0 + srow) * sIn + kk + sseg;
        const unsigned la = (unsigned)(uintptr_t)&As[bufi][srow * 32 + sseg];
        ASYNC_B128(la, ga);
        ASYNC_B128(la + 16u, ga + 8);
        const int n = n0 + srow;
        if (n < Cout) {
            const _Float16* gb = W + (size_t)n * Kdim + kk + sseg;
            const unsigned lb = (unsigned)(uintptr_t)&Bs[bufi][srow * 32 + sseg];
            ASYNC_B128(lb, gb);
            ASYNC_B128(lb + 16u, gb + 8);
        }
    };

    v8f acc[2][4] = {};

    stage(0, 0);                       // prologue: tile 0 -> buffer 0
    int cur = 0;
    for (int kk = 0; kk < Kdim; kk += 32) {
        WAIT_ASYNC0();                 // my DMAs for tile kk done
        __syncthreads();               // everyone's done; other buffer is free
        if (kk + 32 < Kdim) stage(kk + 32, cur ^ 1);

        const _Float16* Ab = &As[cur][0];
        const _Float16* Bb = &Bs[cur][0];

        // A fragments per ISA 16-bit A 16x32 wave32 layout
        HFrag a0, a1;
        const int ar0 = (wm * 32 + mrow) * 32;
        const int ar1 = (wm * 32 + 16 + mrow) * 32;
        a0.h[0] = *(const v8h*)&Ab[ar0 + hsel * 8];
        a0.h[1] = *(const v8h*)&Ab[ar0 + 16 + hsel * 8];
        a1.h[0] = *(const v8h*)&Ab[ar1 + hsel * 8];
        a1.h[1] = *(const v8h*)&Ab[ar1 + 16 + hsel * 8];

#pragma unroll
        for (int j = 0; j < 4; ++j) {
            HFrag bf;
            const int ncol = (wn * 64 + j * 16 + mrow) * 32;
            bf.h[0] = *(const v8h*)&Bb[ncol + hsel * 8];
            bf.h[1] = *(const v8h*)&Bb[ncol + 16 + hsel * 8];
            acc[0][j] = __builtin_amdgcn_wmma_f32_16x16x32_f16(
                false, a0.v, false, bf.v, (short)0, acc[0][j], false, false);
            acc[1][j] = __builtin_amdgcn_wmma_f32_16x16x32_f16(
                false, a1.v, false, bf.v, (short)0, acc[1][j], false, false);
        }
        cur ^= 1;
    }

    // epilogue: C/D layout -> VGPR r: M = r + 8*(lane>=16), N = lane%16
#pragma unroll
    for (int j = 0; j < 4; ++j) {
        const int col = n0 + wn * 64 + j * 16 + mrow;
        if (col < Cout) {
            const float bv = bias[col];
#pragma unroll
            for (int i = 0; i < 2; ++i) {
#pragma unroll
                for (int r = 0; r < 8; ++r) {
                    const int row = t0 + wm * 32 + i * 16 + hsel * 8 + r;
                    out[((size_t)blockIdx.z * T_ + row) * sOut + col] =
                        (OutT)(acc[i][j][r] + bv);
                }
            }
        }
    }
}

// ---------------------------------------------------------------------------
// Weight repack: dst[co][k*Cs + ci] = w[co][ci][k]   (enc, zero-padded ci)
//                dst[co][k*Cs + ci] = w[ci][co][K-1-k] (convT: flip+swap)
// ---------------------------------------------------------------------------
__global__ void k_prep_w(const float* __restrict__ src, _Float16* __restrict__ dst,
                         int Cin, int Cs, int Ksz, int Cout, int transposed)
{
    const int total = Cout * Ksz * Cs;
    for (int idx = blockIdx.x * blockDim.x + threadIdx.x; idx < total;
         idx += gridDim.x * blockDim.x) {
        const int co  = idx / (Ksz * Cs);
        const int rem = idx - co * (Ksz * Cs);
        const int k   = rem / Cs;
        const int ci  = rem - k * Cs;
        float v = 0.f;
        if (ci < Cin) {
            v = transposed ? src[((size_t)ci * Cout + co) * Ksz + (Ksz - 1 - k)]
                           : src[((size_t)co * Cin + ci) * Ksz + k];
        }
        dst[idx] = (_Float16)v;
    }
}

__global__ void k_spk_norm(const float* __restrict__ spk, _Float16* __restrict__ out)
{
    __shared__ float red[128];
    const int r = blockIdx.x, t = threadIdx.x;
    const float v = spk[r * YD_ + t];
    red[t] = v * v;
    __syncthreads();
    for (int o = 64; o > 0; o >>= 1) {
        if (t < o) red[t] += red[t + o];
        __syncthreads();
    }
    out[r * YD_ + t] = (_Float16)(v * rsqrtf(red[0]));
}

__global__ void k_cb_prep(const float* __restrict__ cb, _Float16* __restrict__ cb16,
                          float* __restrict__ cnorm)
{
    __shared__ float red[64];
    const int r = blockIdx.x, t = threadIdx.x;
    const float v = cb[r * ZD_ + t];
    cb16[r * ZD_ + t] = (_Float16)v;
    red[t] = v * v;
    __syncthreads();
    for (int o = 32; o > 0; o >>= 1) {
        if (t < o) red[t] += red[t + o];
        __syncthreads();
    }
    if (t == 0) cnorm[r] = red[0];
}

// x [B,T,80] f32 -> padded f16 [B,T+4,96] (pad rows + cols zeroed)
__global__ void k_pack_x(const float* __restrict__ x, _Float16* __restrict__ dst)
{
    const int total = BN_ * TP_ * 96;
    for (int idx = blockIdx.x * blockDim.x + threadIdx.x; idx < total;
         idx += gridDim.x * blockDim.x) {
        const int b   = idx / (TP_ * 96);
        const int rem = idx - b * (TP_ * 96);
        const int r   = rem / 96;
        const int c   = rem - r * 96;
        float v = 0.f;
        if (r >= 2 && r < T_ + 2 && c < F_)
            v = x[((size_t)b * T_ + (r - 2)) * F_ + c];
        dst[idx] = (_Float16)v;
    }
}

// zero rows {0,1,T+2,T+3} of a padded buffer with given column stride
__global__ void k_zero_pad(_Float16* __restrict__ buf, int stride)
{
    const int total = BN_ * 4 * stride;
    for (int idx = blockIdx.x * blockDim.x + threadIdx.x; idx < total;
         idx += gridDim.x * blockDim.x) {
        const int b   = idx / (4 * stride);
        const int rem = idx - b * (4 * stride);
        const int pr  = rem / stride;
        const int c   = rem - pr * stride;
        const int row = (pr < 2) ? pr : (T_ + pr);
        buf[((size_t)b * TP_ + row) * stride + c] = (_Float16)0.f;
    }
}

// speaker embedding rows into concat columns of a padded buffer
__global__ void k_ye(const int* __restrict__ y, const _Float16* __restrict__ spk16,
                     _Float16* __restrict__ buf, int stride, int colOff)
{
    const int p = blockIdx.x;            // 0 .. B*T-1
    const int b = p / T_, t = p - b * T_;
    const int idx = y[p];
    buf[((size_t)b * TP_ + t + 2) * stride + colOff + threadIdx.x] =
        spk16[idx * YD_ + threadIdx.x];
}

// LayerNorm(channels) + LeakyReLU -> padded f16 buffer
__global__ __launch_bounds__(256)
void k_ln_act(const _Float16* __restrict__ conv, const float* __restrict__ g,
              const float* __restrict__ bt, _Float16* __restrict__ out,
              int Cc, int strideOut)
{
    __shared__ float r1[256], r2[256];
    const int p = blockIdx.x;
    const int b = p / T_, t = p - b * T_;
    const _Float16* row = conv + (size_t)p * Cc;
    float s = 0.f, s2 = 0.f;
    for (int c = threadIdx.x; c < Cc; c += blockDim.x) {
        const float v = (float)row[c];
        s += v; s2 += v * v;
    }
    r1[threadIdx.x] = s; r2[threadIdx.x] = s2;
    __syncthreads();
    for (int o = 128; o > 0; o >>= 1) {
        if (threadIdx.x < o) { r1[threadIdx.x] += r1[threadIdx.x + o];
                               r2[threadIdx.x] += r2[threadIdx.x + o]; }
        __syncthreads();
    }
    const float mean = r1[0] / Cc;
    const float var  = r2[0] / Cc - mean * mean;
    const float rs   = rsqrtf(var + 1e-5f);
    _Float16* orow = out + ((size_t)b * TP_ + t + 2) * strideOut;
    for (int c = threadIdx.x; c < Cc; c += blockDim.x) {
        float v = ((float)row[c] - mean) * rs * g[c] + bt[c];
        v = (v < 0.f) ? v * 0.2f : v;
        orow[c] = (_Float16)v;
    }
}

// LayerNorm(1024) + GLU -> 512 f16 cols of padded buffer
__global__ __launch_bounds__(256)
void k_glu(const _Float16* __restrict__ conv, const float* __restrict__ g,
           const float* __restrict__ bt, _Float16* __restrict__ out, int strideOut)
{
    constexpr int Cc = 2 * C_;
    __shared__ float r1[256], r2[256];
    __shared__ float hs[Cc];
    const int p = blockIdx.x;
    const int b = p / T_, t = p - b * T_;
    const _Float16* row = conv + (size_t)p * Cc;
    float s = 0.f, s2 = 0.f;
    for (int c = threadIdx.x; c < Cc; c += blockDim.x) {
        const float v = (float)row[c];
        s += v; s2 += v * v;
    }
    r1[threadIdx.x] = s; r2[threadIdx.x] = s2;
    __syncthreads();
    for (int o = 128; o > 0; o >>= 1) {
        if (threadIdx.x < o) { r1[threadIdx.x] += r1[threadIdx.x + o];
                               r2[threadIdx.x] += r2[threadIdx.x + o]; }
        __syncthreads();
    }
    const float mean = r1[0] / Cc;
    const float var  = r2[0] / Cc - mean * mean;
    const float rs   = rsqrtf(var + 1e-5f);
    for (int c = threadIdx.x; c < Cc; c += blockDim.x)
        hs[c] = ((float)row[c] - mean) * rs * g[c] + bt[c];
    __syncthreads();
    _Float16* orow = out + ((size_t)b * TP_ + t + 2) * strideOut;
    for (int c = threadIdx.x; c < C_; c += blockDim.x) {
        const float a  = hs[c];
        const float gt = hs[c + C_];
        orow[c] = (_Float16)(a * (1.f / (1.f + expf(-gt))));
    }
}

// VQ: per (b,t) argmin_c ( |c|^2 - 2 z.c ), write codebook[idx] into buffer
__global__ __launch_bounds__(256)
void k_vq(const _Float16* __restrict__ z, const _Float16* __restrict__ cb16,
          const float* __restrict__ cnorm, _Float16* __restrict__ out, int strideOut)
{
    __shared__ _Float16 scb[ZN_ * ZD_];   // 64 KB
    __shared__ float scn[ZN_];
    __shared__ float zsh[8][ZD_];
    for (int i = threadIdx.x; i < ZN_ * ZD_; i += blockDim.x) scb[i] = cb16[i];
    for (int i = threadIdx.x; i < ZN_; i += blockDim.x) scn[i] = cnorm[i];
    __syncthreads();

    const int wave = threadIdx.x >> 5;
    const int lane = threadIdx.x & 31;
    const int total = BN_ * T_;

    for (int p = blockIdx.x * 8 + wave; p < total; p += gridDim.x * 8) {
        const _Float16* zr = z + (size_t)p * ZD_;
        zsh[wave][lane * 2]     = (float)zr[lane * 2];
        zsh[wave][lane * 2 + 1] = (float)zr[lane * 2 + 1];

        float best = 3.4e38f;
        int bi = 0;
        for (int ci = 0; ci < ZN_ / 32; ++ci) {
            const int c = lane + ci * 32;
            const _Float16* crow = &scb[c * ZD_];
            float d = 0.f;
#pragma unroll
            for (int k = 0; k < ZD_; ++k) d += zsh[wave][k] * (float)crow[k];
            d = scn[c] - 2.f * d;
            if (d < best) { best = d; bi = c; }
        }
        for (int m = 16; m >= 1; m >>= 1) {
            const float od = __shfl_xor(best, m, 32);
            const int   oi = __shfl_xor(bi, m, 32);
            if (od < best || (od == best && oi < bi)) { best = od; bi = oi; }
        }
        const int b = p / T_, t = p - b * T_;
        _Float16* orow = out + ((size_t)b * TP_ + t + 2) * strideOut;
        orow[lane * 2]     = scb[bi * ZD_ + lane * 2];
        orow[lane * 2 + 1] = scb[bi * ZD_ + lane * 2 + 1];
    }
}

// ---------------------------------------------------------------------------
extern "C" void kernel_launch(void* const* d_in, const int* in_sizes, int n_in,
                              void* d_out, int out_size, void* d_ws, size_t ws_size,
                              hipStream_t stream)
{
    (void)in_sizes; (void)n_in; (void)out_size; (void)ws_size;

    const float* x        = (const float*)d_in[0];
    const int*   y        = (const int*)d_in[1];
    const float* enc_w1   = (const float*)d_in[2];
    const float* enc_b1   = (const float*)d_in[3];
    const float* enc_g1   = (const float*)d_in[4];
    const float* enc_bt1  = (const float*)d_in[5];
    const float* enc_w2   = (const float*)d_in[6];
    const float* enc_b2   = (const float*)d_in[7];
    const float* enc_g2   = (const float*)d_in[8];
    const float* enc_bt2  = (const float*)d_in[9];
    const float* enc_w3   = (const float*)d_in[10];
    const float* enc_b3   = (const float*)d_in[11];
    const float* enc_g3   = (const float*)d_in[12];
    const float* enc_bt3  = (const float*)d_in[13];
    const float* mlp_w    = (const float*)d_in[14];
    const float* mlp_b    = (const float*)d_in[15];
    const float* codebook = (const float*)d_in[16];
    const float* spk_emb  = (const float*)d_in[17];
    const float* dec_w1   = (const float*)d_in[18];
    const float* dec_b1   = (const float*)d_in[19];
    const float* dec_g1   = (const float*)d_in[20];
    const float* dec_bt1  = (const float*)d_in[21];
    const float* dec_w2   = (const float*)d_in[22];
    const float* dec_b2   = (const float*)d_in[23];
    const float* dec_g2   = (const float*)d_in[24];
    const float* dec_bt2  = (const float*)d_in[25];
    const float* dec_w3   = (const float*)d_in[26];
    const float* dec_b3   = (const float*)d_in[27];
    float*       outp     = (float*)d_out;

    // ---- workspace carve-out (256B aligned) ----
    char* ws = (char*)d_ws;
    size_t off = 0;
    auto take = [&](size_t bytes) -> char* {
        char* p = ws + off;
        off = (off + bytes + 255) & ~(size_t)255;
        return p;
    };
    _Float16* bufA    = (_Float16*)take((size_t)BN_ * TP_ * 640 * 2);
    _Float16* bufB    = (_Float16*)take((size_t)BN_ * TP_ * 640 * 2);
    _Float16* convout = (_Float16*)take((size_t)BN_ * T_ * 1024 * 2);
    _Float16* W_e1 = (_Float16*)take((size_t)512 * 480 * 2);
    _Float16* W_e2 = (_Float16*)take((size_t)512 * 2560 * 2);
    _Float16* W_e3 = (_Float16*)take((size_t)512 * 2560 * 2);
    _Float16* W_ml = (_Float16*)take((size_t)64 * 512 * 2);
    _Float16* W_d1 = (_Float16*)take((size_t)1024 * 960 * 2);
    _Float16* W_d2 = (_Float16*)take((size_t)1024 * 3200 * 2);
    _Float16* W_d3 = (_Float16*)take((size_t)80 * 3200 * 2);
    _Float16* spk16 = (_Float16*)take((size_t)128 * 128 * 2);
    _Float16* cb16  = (_Float16*)take((size_t)ZN_ * ZD_ * 2);
    float*    cnorm = (float*)take((size_t)ZN_ * 4);

    auto gs = [](size_t total) { return dim3((unsigned)((total + 255) / 256)); };

    // ---- weight / constant prep ----
    k_prep_w<<<gs(512u * 480), 256, 0, stream>>>(enc_w1, W_e1, 80, 96, 5, 512, 0);
    k_prep_w<<<gs(512u * 2560), 256, 0, stream>>>(enc_w2, W_e2, 512, 512, 5, 512, 0);
    k_prep_w<<<gs(512u * 2560), 256, 0, stream>>>(enc_w3, W_e3, 512, 512, 5, 512, 0);
    k_prep_w<<<gs(64u * 512), 256, 0, stream>>>(mlp_w, W_ml, 512, 512, 1, 64, 0);
    k_prep_w<<<gs(1024u * 960), 256, 0, stream>>>(dec_w1, W_d1, 192, 192, 5, 1024, 1);
    k_prep_w<<<gs(1024u * 3200), 256, 0, stream>>>(dec_w2, W_d2, 640, 640, 5, 1024, 1);
    k_prep_w<<<gs(80u * 3200), 256, 0, stream>>>(dec_w3, W_d3, 640, 640, 5, 80, 1);
    k_spk_norm<<<128, 128, 0, stream>>>(spk_emb, spk16);
    k_cb_prep<<<ZN_, ZD_, 0, stream>>>(codebook, cb16, cnorm);

    // ---- pack input ----
    k_pack_x<<<gs((size_t)BN_ * TP_ * 96), 256, 0, stream>>>(x, bufA);

    const int rows = BN_ * T_;
    auto conv_h = [&](const _Float16* inBase, int sIn, const _Float16* W,
                      const float* bias, int Kdim, int Cout, _Float16* out) {
        dim3 g(T_ / 128, (Cout + 127) / 128, BN_);
        conv_gemm<_Float16><<<g, 256, 0, stream>>>(
            inBase, (size_t)TP_ * sIn, sIn, W, bias, Kdim, Cout, out, Cout);
    };

    // ---- encoder ----
    conv_h(bufA, 96, W_e1, enc_b1, 480, 512, convout);
    k_zero_pad<<<gs((size_t)BN_ * 4 * 512), 256, 0, stream>>>(bufB, 512);
    k_ln_act<<<rows, 256, 0, stream>>>(convout, enc_g1, enc_bt1, bufB, 512, 512);

    conv_h(bufB, 512, W_e2, enc_b2, 2560, 512, convout);
    k_zero_pad<<<gs((size_t)BN_ * 4 * 512), 256, 0, stream>>>(bufA, 512);
    k_ln_act<<<rows, 256, 0, stream>>>(convout, enc_g2, enc_bt2, bufA, 512, 512);

    conv_h(bufA, 512, W_e3, enc_b3, 2560, 512, convout);
    k_zero_pad<<<gs((size_t)BN_ * 4 * 512), 256, 0, stream>>>(bufB, 512);
    k_ln_act<<<rows, 256, 0, stream>>>(convout, enc_g3, enc_bt3, bufB, 512, 512);

    // ---- 1x1 conv to z (window = row t itself -> base offset +2 rows) ----
    conv_h(bufB + 2 * 512, 512, W_ml, mlp_b, 512, 64, convout);  // z: [B,T,64] f16

    // ---- VQ + speaker concat into dec1 input (stride 192: 64 code + 128 spk) ----
    k_zero_pad<<<gs((size_t)BN_ * 4 * 192), 256, 0, stream>>>(bufA, 192);
    k_vq<<<1024, 256, 0, stream>>>(convout, cb16, cnorm, bufA, 192);
    k_ye<<<rows, 128, 0, stream>>>(y, spk16, bufA, 192, 64);

    // ---- decoder ----
    conv_h(bufA, 192, W_d1, dec_b1, 960, 1024, convout);
    k_zero_pad<<<gs((size_t)BN_ * 4 * 640), 256, 0, stream>>>(bufB, 640);
    k_glu<<<rows, 256, 0, stream>>>(convout, dec_g1, dec_bt1, bufB, 640);
    k_ye<<<rows, 128, 0, stream>>>(y, spk16, bufB, 640, 512);

    conv_h(bufB, 640, W_d2, dec_b2, 3200, 1024, convout);
    k_zero_pad<<<gs((size_t)BN_ * 4 * 640), 256, 0, stream>>>(bufA, 640);
    k_glu<<<rows, 256, 0, stream>>>(convout, dec_g2, dec_bt2, bufA, 640);
    k_ye<<<rows, 128, 0, stream>>>(y, spk16, bufA, 640, 512);

    // ---- final conv straight to f32 output [B,T,80] ----
    {
        dim3 g(T_ / 128, 1, BN_);
        conv_gemm<float><<<g, 256, 0, stream>>>(
            bufA, (size_t)TP_ * 640, 640, W_d3, dec_b3, 3200, 80, outp, 80);
    }
}
